// GIN_34883724378267
// MI455X (gfx1250) — matrix operand order repaired and checked by reading, not compiled, not run on hardware
//
#include <hip/hip_runtime.h>
#include <hip/hip_bf16.h>

#define DH 96
#define NGRP 6          // 96 / 16 output groups
#define NGRAPHS 256

typedef __attribute__((ext_vector_type(16))) __bf16 v16bf;
typedef __attribute__((ext_vector_type(8)))  float  v8f;

// ---------------- small prep kernels ----------------

// Fold BN(eval) + linear bias into per-channel scale/shift:
// y_bn = (x@W^T)*s + ((b - rm)*s + be),  s = g * rsqrt(rv + eps)
__global__ void bn_fold_kernel(const float* __restrict__ g, const float* __restrict__ be,
                               const float* __restrict__ rm, const float* __restrict__ rv,
                               const float* __restrict__ bias,
                               float* __restrict__ scale, float* __restrict__ shift) {
    int i = threadIdx.x;
    if (i < DH) {
        float s = g[i] * rsqrtf(rv[i] + 1e-5f);
        scale[i] = s;
        shift[i] = (bias[i] - rm[i]) * s + be[i];
    }
}

// Pack W [outdim = ngroups*16][K] row-major into WMMA B-fragment layout:
// frag[ng][kc][lane][j] = W[n][k], n = ng*16 + (lane&15),
// k = kc*32 + ((lane&16)?8:0) + (j&7) + ((j&8)?16:0)   (dense bf16 B 32x16 layout)
__global__ void pack_frags_kernel(const float* __restrict__ W, __bf16* __restrict__ out,
                                  int ngroups, int kchunks, int K) {
    int idx = blockIdx.x * blockDim.x + threadIdx.x;
    int total = ngroups * kchunks * 32 * 16;
    if (idx >= total) return;
    int j    = idx & 15;
    int lane = (idx >> 4) & 31;
    int rest = idx >> 9;
    int kc   = rest % kchunks;
    int ng   = rest / kchunks;
    int k = kc * 32 + ((lane & 16) ? 8 : 0) + (j & 7) + ((j & 8) ? 16 : 0);
    int n = ng * 16 + (lane & 15);
    float v = (k < K) ? W[n * K + k] : 0.0f;
    out[idx] = (__bf16)v;
}

__global__ void zerof_kernel(float* __restrict__ p, int n) {
    int i = blockIdx.x * blockDim.x + threadIdx.x;
    if (i < n) p[i] = 0.0f;
}

// ---------------- aggregation (message passing) ----------------

// s1[i][0..6] = x[i][0..6], s1[i][7] = 0  (stride 8 for alignment / K padding)
__global__ void init_s1_kernel(const float* __restrict__ x, float* __restrict__ s1, int n) {
    int i = blockIdx.x * blockDim.x + threadIdx.x;
    if (i < n) {
#pragma unroll
        for (int c = 0; c < 7; ++c) s1[i * 8 + c] = x[i * 7 + c];
        s1[i * 8 + 7] = 0.0f;
    }
}

// edge-parallel scatter for conv1 (7-wide features)
__global__ void scatter_in_kernel(const float* __restrict__ x,
                                  const int* __restrict__ src, const int* __restrict__ dst,
                                  float* __restrict__ s1, int E) {
    int e = blockIdx.x * blockDim.x + threadIdx.x;
    if (e < E) {
        int s = src[e], d = dst[e];
#pragma unroll
        for (int c = 0; c < 7; ++c)
            atomicAdd(&s1[d * 8 + c], x[s * 7 + c]);
    }
}

__global__ void copy_kernel(const float* __restrict__ a, float* __restrict__ b, long long n) {
    long long i = (long long)blockIdx.x * blockDim.x + threadIdx.x;
    if (i < n) b[i] = a[i];
}

// edge x 8-float-chunk parallel scatter for conv2 (96-wide features)
__global__ void scatter_h_kernel(const float* __restrict__ h1,
                                 const int* __restrict__ src, const int* __restrict__ dst,
                                 float* __restrict__ s2, long long total) {
    long long t = (long long)blockIdx.x * blockDim.x + threadIdx.x;
    if (t < total) {
        int e  = (int)(t / 12);
        int c0 = (int)(t % 12) * 8;
        int s = src[e], d = dst[e];
        const float* hs = h1 + (long long)s * DH + c0;
        float* sd = s2 + (long long)d * DH + c0;
#pragma unroll
        for (int i = 0; i < 8; ++i)
            atomicAdd(&sd[i], hs[i]);
    }
}

// ---------------- fused GIN MLP (two WMMA GEMMs + BN + ReLU + pool) ----------------
//
// A-fragment slot for matrix value (m, k):   [ISA 16-bit A 16x32 layout]
//   ki = k & 31, kc = k >> 5
//   lane L = m + 16 * ((ki >> 3) & 1)
//   elem j = (ki & 7) | ((ki >> 1) & 8)
// Consumer reads one contiguous v16bf (32B) per lane per chunk.
template <int KIN, int KCH>
__global__ __launch_bounds__(128)
void mlp_conv_kernel(const float* __restrict__ sin, int n_nodes,
                     const __bf16* __restrict__ fragW1,  // [NGRP][KCH][32][16]
                     const float* __restrict__ scaleA, const float* __restrict__ shiftA,
                     const __bf16* __restrict__ fragW2,  // [NGRP][3][32][16]
                     const float* __restrict__ biasB,
                     float* __restrict__ h_out,          // [N][96] or nullptr
                     float* __restrict__ pool,           // [NGRAPHS][96]
                     const int* __restrict__ batch) {
    __shared__ __align__(32) __bf16 s_a1[4][KCH][32][16];  // GEMM1 A fragments
    __shared__ __align__(32) __bf16 s_a2[4][3][32][16];    // GEMM2 A fragments

    const int wave  = threadIdx.x >> 5;
    const int lane  = threadIdx.x & 31;
    const int tile  = blockIdx.x * 4 + wave;
    const int node0 = tile * 16;

    __bf16* a1 = &s_a1[wave][0][0][0];
    __bf16* a2 = &s_a2[wave][0][0][0];

    if (KIN < KCH * 32) {
        // zero-fill padded K slots (vectorized)
        uint4* za = (uint4*)a1;     // KCH*512 bf16 = KCH*64 uint4
        for (int t = lane; t < KCH * 64; t += 32) za[t] = uint4{0, 0, 0, 0};
    }
    __syncthreads();

    // cooperative coalesced tile load -> convert -> scatter into fragment slots
    for (int t = lane; t < 16 * KIN; t += 32) {
        int m = t / KIN, k = t % KIN;
        int node = node0 + m;
        float v = (node < n_nodes) ? sin[(long long)node0 * KIN + t] : 0.0f;
        int kc = k >> 5, ki = k & 31;
        int L = m + (((ki >> 3) & 1) << 4);
        int j = (ki & 7) | ((ki >> 1) & 8);
        a1[(kc * 32 + L) * 16 + j] = (__bf16)v;
    }
    __syncthreads();

    const int row = lane & 15;
    const int mb  = (lane & 16) ? 8 : 0;   // C-fragment M base

    // ---- GEMM 1: [16,KIN] x [KIN,96], BN-affine + ReLU -> GEMM2 A fragments ----
#pragma unroll
    for (int ng = 0; ng < NGRP; ++ng) {
        v8f acc = {};
#pragma unroll
        for (int kc = 0; kc < KCH; ++kc) {
            v16bf a = *(const v16bf*)(a1 + (kc * 32 + lane) * 16);
            v16bf b = *(const v16bf*)(fragW1 + ((ng * KCH + kc) * 32 + lane) * 16);
            acc = __builtin_amdgcn_wmma_f32_16x16x32_bf16(
                false, a, false, b, (short)0, acc, false, false);
        }
        int n = ng * 16 + row;                // this lane's output column
        float sc = scaleA[n], sh = shiftA[n];
        // destination fragment coordinates for k = n
        int kc2 = n >> 5, ki = n & 31;
        int Lh  = ((ki >> 3) & 1) << 4;
        int j2  = (ki & 7) | ((ki >> 1) & 8);
        __bf16* dstf = a2 + (kc2 * 32 + Lh) * 16 + j2;
#pragma unroll
        for (int r = 0; r < 8; ++r) {
            int m = mb + r;
            float v = fmaxf(acc[r] * sc + sh, 0.0f);
            dstf[m * 16] = (__bf16)v;
        }
    }
    __syncthreads();

    // hoist per-row metadata out of the unrolled output loops
    int   bat8[8];
#pragma unroll
    for (int r = 0; r < 8; ++r) {
        int node = node0 + mb + r;
        bat8[r] = (node < n_nodes) ? batch[node] : -1;
    }

    // ---- GEMM 2: [16,96] x [96,96], bias + ReLU, store + pool ----
#pragma unroll
    for (int ng = 0; ng < NGRP; ++ng) {
        v8f acc = {};
#pragma unroll
        for (int kc = 0; kc < 3; ++kc) {
            v16bf a = *(const v16bf*)(a2 + (kc * 32 + lane) * 16);
            v16bf b = *(const v16bf*)(fragW2 + ((ng * 3 + kc) * 32 + lane) * 16);
            acc = __builtin_amdgcn_wmma_f32_16x16x32_bf16(
                false, a, false, b, (short)0, acc, false, false);
        }
        int n = ng * 16 + row;
        float bb = biasB[n];
#pragma unroll
        for (int r = 0; r < 8; ++r) {
            if (bat8[r] >= 0) {
                int node = node0 + mb + r;
                float v = fmaxf(acc[r] + bb, 0.0f);
                if (h_out) h_out[(long long)node * DH + n] = v;
                atomicAdd(&pool[bat8[r] * DH + n], v);
            }
        }
    }
}

// ---------------- head: Linear(192,192)+ReLU, Linear(192,2), log_softmax ----------------
__global__ __launch_bounds__(192)
void head_kernel(const float* __restrict__ g1p, const float* __restrict__ g2p,
                 const float* __restrict__ wl1, const float* __restrict__ bl1,
                 const float* __restrict__ wl2, const float* __restrict__ bl2,
                 float* __restrict__ out) {
    __shared__ float hcat[2 * DH];
    __shared__ float y1[2 * DH];
    __shared__ float y2[2];
    int g = blockIdx.x;
    int t = threadIdx.x;
    hcat[t] = (t < DH) ? g1p[g * DH + t] : g2p[g * DH + (t - DH)];
    __syncthreads();
    float acc = bl1[t];
    for (int k = 0; k < 2 * DH; ++k) acc += wl1[t * 2 * DH + k] * hcat[k];
    y1[t] = fmaxf(acc, 0.0f);
    __syncthreads();
    if (t < 2) {
        float a = bl2[t];
        for (int k = 0; k < 2 * DH; ++k) a += wl2[t * 2 * DH + k] * y1[k];
        y2[t] = a;
    }
    __syncthreads();
    if (t == 0) {
        float a0 = y2[0], a1 = y2[1];
        float mx = fmaxf(a0, a1);
        float lse = mx + logf(expf(a0 - mx) + expf(a1 - mx));
        out[g * 2 + 0] = a0 - lse;
        out[g * 2 + 1] = a1 - lse;
    }
}

// ---------------- launch ----------------

extern "C" void kernel_launch(void* const* d_in, const int* in_sizes, int n_in,
                              void* d_out, int out_size, void* d_ws, size_t ws_size,
                              hipStream_t stream) {
    const float* x   = (const float*)d_in[0];
    const int*   ei  = (const int*)d_in[1];
    const int*   bat = (const int*)d_in[2];
    const float* w1a = (const float*)d_in[3];
    const float* b1a = (const float*)d_in[4];
    const float* g1  = (const float*)d_in[5];
    const float* be1 = (const float*)d_in[6];
    const float* rm1 = (const float*)d_in[7];
    const float* rv1 = (const float*)d_in[8];
    const float* w1b = (const float*)d_in[9];
    const float* b1b = (const float*)d_in[10];
    const float* w2a = (const float*)d_in[11];
    const float* b2a = (const float*)d_in[12];
    const float* g2  = (const float*)d_in[13];
    const float* be2 = (const float*)d_in[14];
    const float* rm2 = (const float*)d_in[15];
    const float* rv2 = (const float*)d_in[16];
    const float* w2b = (const float*)d_in[17];
    const float* b2b = (const float*)d_in[18];
    const float* wl1 = (const float*)d_in[19];
    const float* bl1 = (const float*)d_in[20];
    const float* wl2 = (const float*)d_in[21];
    const float* bl2 = (const float*)d_in[22];

    const int N = in_sizes[0] / 7;
    const int E = in_sizes[1] / 2;
    const int* src = ei;
    const int* dst = ei + E;

    auto alignup = [](size_t v) { return (v + 255) & ~(size_t)255; };
    char* wsp = (char*)d_ws;
    size_t off = 0;
    float* s1    = (float*)(wsp + off); off += alignup((size_t)N * 8 * 4);
    float* h1    = (float*)(wsp + off); off += alignup((size_t)N * DH * 4);
    float* s2    = (float*)(wsp + off); off += alignup((size_t)N * DH * 4);
    float* pools = (float*)(wsp + off); off += alignup((size_t)2 * NGRAPHS * DH * 4);
    float* g1p = pools;
    float* g2p = pools + NGRAPHS * DH;
    float* c1s   = (float*)(wsp + off); off += alignup(4 * DH * 4);
    float* c1t = c1s + DH;
    float* c2s = c1s + 2 * DH;
    float* c2t = c1s + 3 * DH;
    __bf16* f_w1a = (__bf16*)(wsp + off); off += alignup((size_t)NGRP * 1 * 32 * 16 * 2);
    __bf16* f_w1b = (__bf16*)(wsp + off); off += alignup((size_t)NGRP * 3 * 32 * 16 * 2);
    __bf16* f_w2a = (__bf16*)(wsp + off); off += alignup((size_t)NGRP * 3 * 32 * 16 * 2);
    __bf16* f_w2b = (__bf16*)(wsp + off); off += alignup((size_t)NGRP * 3 * 32 * 16 * 2);
    (void)ws_size; (void)n_in; (void)out_size;

    // 1. fold BN + pack weight fragments (tiny)
    bn_fold_kernel<<<1, DH, 0, stream>>>(g1, be1, rm1, rv1, b1a, c1s, c1t);
    bn_fold_kernel<<<1, DH, 0, stream>>>(g2, be2, rm2, rv2, b2a, c2s, c2t);
    {
        int tot1 = NGRP * 1 * 512, tot3 = NGRP * 3 * 512;
        pack_frags_kernel<<<(tot1 + 255) / 256, 256, 0, stream>>>(w1a, f_w1a, NGRP, 1, 7);
        pack_frags_kernel<<<(tot3 + 255) / 256, 256, 0, stream>>>(w1b, f_w1b, NGRP, 3, DH);
        pack_frags_kernel<<<(tot3 + 255) / 256, 256, 0, stream>>>(w2a, f_w2a, NGRP, 3, DH);
        pack_frags_kernel<<<(tot3 + 255) / 256, 256, 0, stream>>>(w2b, f_w2b, NGRP, 3, DH);
    }

    // 2. zero graph pools (accumulated atomically every call)
    {
        int n = 2 * NGRAPHS * DH;
        zerof_kernel<<<(n + 255) / 256, 256, 0, stream>>>(pools, n);
    }

    // 3. conv1 aggregation: s1 = x + scatter_add(x[src] -> dst)
    init_s1_kernel<<<(N + 255) / 256, 256, 0, stream>>>(x, s1, N);
    scatter_in_kernel<<<(E + 255) / 256, 256, 0, stream>>>(x, src, dst, s1, E);

    // 4. conv1 MLP (WMMA) -> h1, pool into g1p
    const int tiles  = (N + 15) / 16;
    const int blocks = (tiles + 3) / 4;
    mlp_conv_kernel<8, 1><<<blocks, 128, 0, stream>>>(
        s1, N, f_w1a, c1s, c1t, f_w1b, b1b, h1, g1p, bat);

    // 5. conv2 aggregation: s2 = h1 + scatter_add(h1[src] -> dst)
    {
        long long n = (long long)N * DH;
        copy_kernel<<<(int)((n + 255) / 256), 256, 0, stream>>>(h1, s2, n);
        long long tot = (long long)E * 12;
        scatter_h_kernel<<<(int)((tot + 255) / 256), 256, 0, stream>>>(h1, src, dst, s2, tot);
    }

    // 6. conv2 MLP (WMMA) -> pool into g2p (h2 itself is not needed)
    mlp_conv_kernel<DH, 3><<<blocks, 128, 0, stream>>>(
        s2, N, f_w2a, c2s, c2t, f_w2b, b2b, (float*)nullptr, g2p, bat);

    // 7. head: concat pools -> Linear+ReLU -> Linear -> log_softmax
    head_kernel<<<NGRAPHS, 192, 0, stream>>>(g1p, g2p, wl1, bl1, wl2, bl2, (float*)d_out);
}
